// CausalSelfAttention_77206332113053
// MI455X (gfx1250) — matrix operand order repaired and checked by reading, not compiled, not run on hardware
//
#include <hip/hip_runtime.h>
#include <hip/hip_bf16.h>

// CDNA5 (gfx1250) wave32 WMMA implementation of causal self-attention.
// Precision path: fp32 inputs -> bf16 operands, fp32 WMMA accumulation.
// GEMMs and attention use block-cooperative double-buffered LDS staging.

typedef __attribute__((ext_vector_type(16))) __bf16          v16bf;
typedef __attribute__((ext_vector_type(16))) unsigned short  v16u;
typedef __attribute__((ext_vector_type(8)))  unsigned short  u16x8;
typedef __attribute__((ext_vector_type(4)))  unsigned short  u16x4;
typedef __attribute__((ext_vector_type(4)))  float           f32x4;
typedef __attribute__((ext_vector_type(8)))  float           v8f;

#define TSEQ   2048
#define CDIM   2048
#define HEADS  16
#define DHEAD  128
#define MROWS  4096   // B*T
#define SCALE  0.08838834764831845f  // 1/sqrt(128)
#define KCH     64    // GEMM K chunk per barrier interval
#define BSTRIDE 72    // GEMM LDS column stride (elements): 144B -> conflict-free
#define BUFELTS (64 * BSTRIDE)
#define KSTR    136   // attn K-tile column stride (elems): 272B -> conflict-free
#define VSTR    40    // attn V-tile column stride (elems): 80B  -> conflict-free
#define KBUFE   (32 * KSTR)    // 4352 elems per K buffer
#define VBUFE   (128 * VSTR)   // 5120 elems per V buffer

__device__ __forceinline__ unsigned short f2bfbits(float f) {
  unsigned int u = __builtin_bit_cast(unsigned int, f);
  u += 0x7FFFu + ((u >> 16) & 1u);           // round-to-nearest-even
  return (unsigned short)(u >> 16);
}

__device__ __forceinline__ v16bf make_frag(u16x8 r0, u16x8 r1) {
  v16u w = __builtin_shufflevector(r0, r1, 0,1,2,3,4,5,6,7,8,9,10,11,12,13,14,15);
  return __builtin_bit_cast(v16bf, w);
}

// A-matrix fragment (16x32 bf16, M x K). ISA: lanes 0-15 row=lane, K={0..7,16..23};
// lanes 16-31 row=lane-16, K={8..15,24..31}. rowbase = &A[row][0] for this lane.
__device__ __forceinline__ v16bf load_fragA(const unsigned short* rowbase, int k0) {
  const int h8 = ((threadIdx.x >> 4) & 1) << 3;
  u16x8 r0 = *reinterpret_cast<const u16x8*>(rowbase + k0 + h8);
  u16x8 r1 = *reinterpret_cast<const u16x8*>(rowbase + k0 + 16 + h8);
  return make_frag(r0, r1);
}

// B-matrix fragment (32x16 bf16, K x N). lanes 0-15 col=lane, K={0..7,16..23};
// lanes 16-31 col=lane-16, K={8..15,24..31} (matches the A-fragment order).
__device__ __forceinline__ v16bf load_fragB(const unsigned short* colbase, int k0) {
  const int h16 = ((threadIdx.x >> 4) & 1) << 3;
  u16x8 r0 = *reinterpret_cast<const u16x8*>(colbase + k0 + h16);
  u16x8 r1 = *reinterpret_cast<const u16x8*>(colbase + k0 + h16 + 16);
  return make_frag(r0, r1);
}

#define WMMA_BF16(A,B,C) \
  __builtin_amdgcn_wmma_f32_16x16x32_bf16(false,(A),false,(B),(short)0,(C),false,false)

// ---------------------------------------------------------------- fp32 -> bf16
__global__ void cvt_bf16_kernel(const float* __restrict__ src,
                                unsigned short* __restrict__ dst, int n4) {
  int i = blockIdx.x * blockDim.x + threadIdx.x;
  int stride = gridDim.x * blockDim.x;
  const f32x4* s4 = reinterpret_cast<const f32x4*>(src);
  u16x4* d4 = reinterpret_cast<u16x4*>(dst);
  for (; i < n4; i += stride) {
    f32x4 v = s4[i];
    u16x4 o;
    o.x = f2bfbits(v.x); o.y = f2bfbits(v.y);
    o.z = f2bfbits(v.z); o.w = f2bfbits(v.w);
    d4[i] = o;
  }
}

// ---------------------------------------------------------------- GEMM (+RoPE)
// C[m,n] = sum_k A[m,k] * W[n,k]   (A: MROWS x CDIM bf16, W: CDIM x CDIM bf16)
// Block tile: 128(M) x 64(N), 8 waves, each wave 16(M) x 64(N).
// Weight tile for each 64-K chunk staged in double-buffered LDS (8x reuse).
// mode 0: Q -> rope bf16 [B,H,T,D] | 1: K -> rope bf16 [B,H,T,D]
// mode 2: V -> bf16 transposed [B,H,D,T] | 3: out -> fp32 [m,n]
__global__ void gemm_wmma_kernel(const unsigned short* __restrict__ A,
                                 const unsigned short* __restrict__ W,
                                 const float* __restrict__ cosp,
                                 const float* __restrict__ sinp,
                                 unsigned short* __restrict__ obf,
                                 float* __restrict__ ofp,
                                 int mode) {
  __shared__ alignas(16) unsigned short bs[2 * BUFELTS];  // 2 x 9KB

  const int tid  = threadIdx.x;
  const int lane = tid & 31;
  const int idx  = lane & 15;
  const int halfR = (lane >> 4) << 3;
  const int wv   = tid >> 5;                 // wave in block: 0..7

  const int bm0 = (blockIdx.x >> 5) << 7;    // 128-row block tile
  const int bn0 = (blockIdx.x & 31) << 6;    // 64-col block tile
  const int m0  = bm0 + (wv << 4);

  // staging: 256 threads move the 64x64 bf16 weight tile (8KB) per chunk
  const int sr = tid >> 2, sseg = tid & 3;   // row 0..63, 16B segment 0..3
  const unsigned short* wsrc = W + (size_t)(bn0 + sr) * CDIM + sseg * 8;
  const int soff = sr * BSTRIDE + sseg * 8;  // integer LDS offset (keeps ds_store)

  const unsigned short* arow = A + (size_t)(m0 + idx) * CDIM;

  v8f acc[4];
#pragma unroll
  for (int j = 0; j < 4; ++j) acc[j] = (v8f)(0.0f);

  // prologue: stage chunk 0, load A frags for chunk 0
  {
    u16x8 g0 = *reinterpret_cast<const u16x8*>(wsrc);
    u16x8 g1 = *reinterpret_cast<const u16x8*>(wsrc + 32);
    *reinterpret_cast<u16x8*>(&bs[soff])      = g0;
    *reinterpret_cast<u16x8*>(&bs[soff + 32]) = g1;
  }
  v16bf ca0 = load_fragA(arow, 0);
  v16bf ca1 = load_fragA(arow, 32);
  asm volatile("s_wait_dscnt 0" ::: "memory");
  __syncthreads();

  int buf = 0;
  for (int kc = KCH; kc < CDIM; kc += KCH) {
    // stage next chunk into the other buffer (integer offset -> ds_store_b128)
    {
      u16x8 g0 = *reinterpret_cast<const u16x8*>(wsrc + kc);
      u16x8 g1 = *reinterpret_cast<const u16x8*>(wsrc + kc + 32);
      const int doff = (buf ^ 1) * BUFELTS + soff;
      *reinterpret_cast<u16x8*>(&bs[doff])      = g0;
      *reinterpret_cast<u16x8*>(&bs[doff + 32]) = g1;
    }
    v16bf na0 = load_fragA(arow, kc);
    v16bf na1 = load_fragA(arow, kc + 32);

    // compute current chunk: batch the 4 B-fragment LDS loads per sub-chunk
    const unsigned short* bb = &bs[buf * BUFELTS];
#pragma unroll
    for (int s = 0; s < 2; ++s) {
      const v16bf cva = (s == 0) ? ca0 : ca1;
      v16bf b0 = load_fragB(bb + (size_t)( 0 + idx) * BSTRIDE, s * 32);
      v16bf b1 = load_fragB(bb + (size_t)(16 + idx) * BSTRIDE, s * 32);
      v16bf b2 = load_fragB(bb + (size_t)(32 + idx) * BSTRIDE, s * 32);
      v16bf b3 = load_fragB(bb + (size_t)(48 + idx) * BSTRIDE, s * 32);
      acc[0] = WMMA_BF16(cva, b0, acc[0]);
      acc[1] = WMMA_BF16(cva, b1, acc[1]);
      acc[2] = WMMA_BF16(cva, b2, acc[2]);
      acc[3] = WMMA_BF16(cva, b3, acc[3]);
    }
    ca0 = na0; ca1 = na1;
    asm volatile("s_wait_dscnt 0" ::: "memory");
    __syncthreads();
    buf ^= 1;
  }
  // epilogue chunk
  {
    const unsigned short* bb = &bs[buf * BUFELTS];
#pragma unroll
    for (int s = 0; s < 2; ++s) {
      const v16bf cva = (s == 0) ? ca0 : ca1;
      v16bf b0 = load_fragB(bb + (size_t)( 0 + idx) * BSTRIDE, s * 32);
      v16bf b1 = load_fragB(bb + (size_t)(16 + idx) * BSTRIDE, s * 32);
      v16bf b2 = load_fragB(bb + (size_t)(32 + idx) * BSTRIDE, s * 32);
      v16bf b3 = load_fragB(bb + (size_t)(48 + idx) * BSTRIDE, s * 32);
      acc[0] = WMMA_BF16(cva, b0, acc[0]);
      acc[1] = WMMA_BF16(cva, b1, acc[1]);
      acc[2] = WMMA_BF16(cva, b2, acc[2]);
      acc[3] = WMMA_BF16(cva, b3, acc[3]);
    }
  }

#pragma unroll
  for (int j = 0; j < 4; ++j) {
    const int n = bn0 + (j << 4) + idx;
#pragma unroll
    for (int e = 0; e < 8; ++e) {
      const int m = m0 + e + halfR;
      float v = acc[j][e];
      if (mode == 3) {
        ofp[(size_t)m * CDIM + n] = v;
      } else {
        const int h = n >> 7, d = n & 127;
        const int b = m >> 11, t = m & 2047;
        if (mode == 2) {
          // V transposed: [B,H,D,T]
          obf[(((size_t)(b * HEADS + h) * DHEAD + d) * TSEQ) + t] = f2bfbits(v);
        } else {
          // RoPE: pairs (2j,2j+1) live in adjacent lanes of the C layout
          float p  = __shfl_xor(v, 1, 32);
          const int jj = d >> 1;
          float c = cosp[t * (DHEAD / 2) + jj];
          float s = sinp[t * (DHEAD / 2) + jj];
          float r = ((lane & 1) == 0) ? (v * c - p * s)   // even: x0*c - x1*s
                                      : (p * s + v * c);  // odd : x0*s + x1*c
          obf[(((size_t)(b * HEADS + h) * TSEQ + t) * DHEAD) + d] = f2bfbits(r);
        }
      }
    }
  }
}

// ---------------------------------------------------------------- flash attention
// Block = 128 q-rows of one (b,h); 8 waves x 16 rows. K/V tiles for each 32-k
// step staged cooperatively into double-buffered LDS (8x reuse of K/V bytes).
// Q:[BH,T,D], K:[BH,T,D], V:[BH,D,T] bf16. Output bf16 [m=b*T+t, n=h*D+d].
__global__ void attn_wmma_kernel(const unsigned short* __restrict__ qg,
                                 const unsigned short* __restrict__ kg,
                                 const unsigned short* __restrict__ vg,
                                 unsigned short* __restrict__ obf) {
  __shared__ alignas(16) unsigned short kbufA[2 * KBUFE];  // 2 x 8.5KB
  __shared__ alignas(16) unsigned short vbufA[2 * VBUFE];  // 2 x 10KB
  __shared__ alignas(16) unsigned short plds[8 * 512];     // P staging per wave

  const int tid  = threadIdx.x;
  const int lane = tid & 31;
  const int idx  = lane & 15;
  const int halfR = (lane >> 4) << 3;
  const int wv   = tid >> 5;

  const int bh   = blockIdx.x >> 4;            // 0..31
  const int qblk = blockIdx.x & 15;            // 128-row q block
  const int q0   = (qblk << 7) + (wv << 4);    // this wave's 16 q rows
  const int kmaxB = (qblk << 7) + 127;         // block causal bound
  const int kmaxW = q0 + 15;                   // wave causal bound

  // K-tile staging: 32 rows x 128 d (8KB); thread -> row tid>>3, 32B at d=(tid&7)*16
  const int krow = tid >> 3, kd0 = (tid & 7) << 4;
  const unsigned short* kgsrc = kg + ((size_t)bh * TSEQ + krow) * DHEAD + kd0;
  const int ksoff = krow * KSTR + kd0;
  // V-tile staging: 128 d-cols x 32 t (8KB); thread -> col tid>>1, 32B at t=(tid&1)*16
  const int vd = tid >> 1, vt0 = (tid & 1) << 4;
  const unsigned short* vgsrc = vg + ((size_t)bh * DHEAD + vd) * TSEQ + vt0;
  const int vsoff = vd * VSTR + vt0;

  // Q fragments held in registers (16 rows x 128 d)
  const unsigned short* qrow = qg + ((size_t)bh * TSEQ + q0 + idx) * DHEAD;
  v16bf qa[4];
#pragma unroll
  for (int c = 0; c < 4; ++c) qa[c] = load_fragA(qrow, c * 32);

  v8f o[8];
  float mr[8], lr[8];
#pragma unroll
  for (int e = 0; e < 8; ++e) { o[e] = (v8f)(0.0f); mr[e] = -1e30f; lr[e] = 0.0f; }

  // prologue: stage k-tile 0
  {
    u16x8 a0 = *reinterpret_cast<const u16x8*>(kgsrc);
    u16x8 a1 = *reinterpret_cast<const u16x8*>(kgsrc + 8);
    u16x8 b0 = *reinterpret_cast<const u16x8*>(vgsrc);
    u16x8 b1 = *reinterpret_cast<const u16x8*>(vgsrc + 8);
    *reinterpret_cast<u16x8*>(&kbufA[ksoff])     = a0;
    *reinterpret_cast<u16x8*>(&kbufA[ksoff + 8]) = a1;
    *reinterpret_cast<u16x8*>(&vbufA[vsoff])     = b0;
    *reinterpret_cast<u16x8*>(&vbufA[vsoff + 8]) = b1;
  }
  asm volatile("s_wait_dscnt 0" ::: "memory");
  __syncthreads();

  int buf = 0;
  for (int kb = 0; kb <= kmaxB; kb += 32) {
    // stage next tile into the other buffer
    if (kb + 32 <= kmaxB) {
      u16x8 a0 = *reinterpret_cast<const u16x8*>(kgsrc + (size_t)(kb + 32) * DHEAD);
      u16x8 a1 = *reinterpret_cast<const u16x8*>(kgsrc + (size_t)(kb + 32) * DHEAD + 8);
      u16x8 b0 = *reinterpret_cast<const u16x8*>(vgsrc + kb + 32);
      u16x8 b1 = *reinterpret_cast<const u16x8*>(vgsrc + kb + 40);
      const int kd = (buf ^ 1) * KBUFE + ksoff;
      const int vdo = (buf ^ 1) * VBUFE + vsoff;
      *reinterpret_cast<u16x8*>(&kbufA[kd])      = a0;
      *reinterpret_cast<u16x8*>(&kbufA[kd + 8])  = a1;
      *reinterpret_cast<u16x8*>(&vbufA[vdo])     = b0;
      *reinterpret_cast<u16x8*>(&vbufA[vdo + 8]) = b1;
    }

    if (kb <= kmaxW) {   // wave-uniform causal skip (EXEC stays all-ones)
      const unsigned short* kt = &kbufA[buf * KBUFE];
      const unsigned short* vt = &vbufA[buf * VBUFE];

      // S = Q K^T : two 16x16 column tiles from the LDS K-tile
      v8f s0 = (v8f)(0.0f), s1 = (v8f)(0.0f);
#pragma unroll
      for (int c = 0; c < 4; ++c) {
        s0 = WMMA_BF16(qa[c], load_fragB(kt + (size_t)( 0 + idx) * KSTR, c * 32), s0);
        s1 = WMMA_BF16(qa[c], load_fragB(kt + (size_t)(16 + idx) * KSTR, c * 32), s1);
      }

      // mask + online softmax, stage P (bf16) through LDS
#pragma unroll
      for (int e = 0; e < 8; ++e) {
        const int row = q0 + e + halfR;
        float a0 = s0[e] * SCALE; if (kb + idx      > row) a0 = -1e30f;
        float a1 = s1[e] * SCALE; if (kb + 16 + idx > row) a1 = -1e30f;
        float mx = fmaxf(a0, a1);
#pragma unroll
        for (int off = 1; off < 16; off <<= 1) mx = fmaxf(mx, __shfl_xor(mx, off, 32));
        const float mn = fmaxf(mr[e], mx);
        const float al = __expf(mr[e] - mn);
        const float p0 = __expf(a0 - mn);
        const float p1 = __expf(a1 - mn);
        float rs = p0 + p1;
#pragma unroll
        for (int off = 1; off < 16; off <<= 1) rs += __shfl_xor(rs, off, 32);
        lr[e] = lr[e] * al + rs;
        mr[e] = mn;
#pragma unroll
        for (int dt = 0; dt < 8; ++dt) o[dt][e] *= al;

        const int r = e + halfR;
        plds[wv * 512 + r * 32 + idx]      = f2bfbits(p0);
        plds[wv * 512 + r * 32 + 16 + idx] = f2bfbits(p1);
      }
      asm volatile("s_wait_dscnt 0" ::: "memory");

      // P (16x32) as A-fragment from LDS, O += P * V (V from LDS tile)
      v16bf pa = load_fragA(plds + wv * 512 + idx * 32, 0);
#pragma unroll
      for (int dt = 0; dt < 8; ++dt) {
        v16bf vb = load_fragB(vt + (size_t)((dt << 4) + idx) * VSTR, 0);
        o[dt] = WMMA_BF16(pa, vb, o[dt]);
      }
    }

    asm volatile("s_wait_dscnt 0" ::: "memory");
    __syncthreads();
    buf ^= 1;
  }

  // normalize and store bf16 for the output projection
  const int b = bh >> 4, h = bh & 15;
#pragma unroll
  for (int e = 0; e < 8; ++e) {
    const float inv = 1.0f / lr[e];
    const int t = q0 + e + halfR;
    const size_t m = (size_t)b * TSEQ + t;
#pragma unroll
    for (int dt = 0; dt < 8; ++dt) {
      const int n = h * DHEAD + (dt << 4) + idx;
      obf[m * CDIM + n] = f2bfbits(o[dt][e] * inv);
    }
  }
}

// ---------------------------------------------------------------- launch
extern "C" void kernel_launch(void* const* d_in, const int* in_sizes, int n_in,
                              void* d_out, int out_size, void* d_ws, size_t ws_size,
                              hipStream_t stream) {
  const float* x  = (const float*)d_in[0];
  const float* fc = (const float*)d_in[1];
  const float* fs = (const float*)d_in[2];
  const float* wq = (const float*)d_in[3];
  const float* wk = (const float*)d_in[4];
  const float* wv = (const float*)d_in[5];
  const float* wo = (const float*)d_in[6];

  char* ws = (char*)d_ws;
  const size_t MB = 1u << 20;
  unsigned short* xbf = (unsigned short*)(ws +   0 * MB);  // 16 MB
  unsigned short* wqb = (unsigned short*)(ws +  16 * MB);  //  8 MB
  unsigned short* wkb = (unsigned short*)(ws +  24 * MB);
  unsigned short* wvb = (unsigned short*)(ws +  32 * MB);
  unsigned short* wob = (unsigned short*)(ws +  40 * MB);
  unsigned short* qws = (unsigned short*)(ws +  48 * MB);  // 16 MB [B,H,T,D]
  unsigned short* kws = (unsigned short*)(ws +  64 * MB);  // 16 MB [B,H,T,D]
  unsigned short* vws = (unsigned short*)(ws +  80 * MB);  // 16 MB [B,H,D,T]
  unsigned short* aws = (unsigned short*)(ws +  96 * MB);  // 16 MB attn out bf16

  cvt_bf16_kernel<<<1024, 256, 0, stream>>>(x,  xbf, (MROWS * CDIM) / 4);
  cvt_bf16_kernel<<<512,  256, 0, stream>>>(wq, wqb, (CDIM * CDIM) / 4);
  cvt_bf16_kernel<<<512,  256, 0, stream>>>(wk, wkb, (CDIM * CDIM) / 4);
  cvt_bf16_kernel<<<512,  256, 0, stream>>>(wv, wvb, (CDIM * CDIM) / 4);
  cvt_bf16_kernel<<<512,  256, 0, stream>>>(wo, wob, (CDIM * CDIM) / 4);

  // (M/128)*(N/64) = 32*32 = 1024 blocks x 256 threads
  gemm_wmma_kernel<<<1024, 256, 0, stream>>>(xbf, wqb, fc, fs, qws, nullptr, 0);
  gemm_wmma_kernel<<<1024, 256, 0, stream>>>(xbf, wkb, fc, fs, kws, nullptr, 1);
  gemm_wmma_kernel<<<1024, 256, 0, stream>>>(xbf, wvb, nullptr, nullptr, vws, nullptr, 2);

  // B*H*(T/128) = 512 blocks x 256 threads
  attn_wmma_kernel<<<512, 256, 0, stream>>>(qws, kws, vws, aws);

  gemm_wmma_kernel<<<1024, 256, 0, stream>>>(aws, wob, nullptr, nullptr, nullptr,
                                             (float*)d_out, 3);
}
